// GroupAttention_4140348473870
// MI455X (gfx1250) — matrix operand order repaired and checked by reading, not compiled, not run on hardware
//
#include <hip/hip_runtime.h>

// GroupAttention for MI455X (gfx1250), wave32 + WMMA bf16.
//   B=4, S=4096, D=256, H=10, G=4, C=1024, HD=2560
// Pipeline:
//   proj_kernel  : Qbf[h,b,s,e] = x@Wq[h], Kbf = x@Wk[h]   (f32 in -> bf16 out)
//   attn_kernel  : flash softmax(Q Kroll^T) Kroll -> cat[b,s,h*D+d] (bf16)
//   out_kernel   : d_out[b,s,d] = cat @ Wo^T                (f32 out)
// Workspace: 3 * H*B*S*D * 2 bytes = 251,658,240 bytes.

#define DEV static __device__ __forceinline__

typedef __bf16 bf16_t;
typedef __attribute__((ext_vector_type(8)))  __bf16 v8bf;
typedef __attribute__((ext_vector_type(16))) __bf16 v16bf;
typedef __attribute__((ext_vector_type(8)))  float  v8f;
typedef __attribute__((ext_vector_type(4)))  float  v4f;

union BF16x16 { v16bf v; v8bf h[2]; };

DEV v8f vzero8() {
  v8f z;
#pragma unroll
  for (int i = 0; i < 8; ++i) z[i] = 0.0f;
  return z;
}

DEV v8f wmma_bf16(v16bf a, v16bf b, v8f c) {
  // D(f32 16x16) = A(bf16 16x32) * B(bf16 32x16) + C
  return __builtin_amdgcn_wmma_f32_16x16x32_bf16(
      /*neg_a=*/false, a, /*neg_b=*/false, b,
      /*c_mod=*/(short)0, c, /*reuse_a=*/false, /*reuse_b=*/false);
}

// A fragment 16x32 bf16 (contiguous k order, matching B loaders):
//   lane L: m = L&15, k = koff + j
//   koff = (L>=16) ? 16 : 0.  Two contiguous 8-element (16B) loads per lane.
DEV v16bf load_a16x32(const bf16_t* base, int stride, int lane) {
  const int m = lane & 15;
  const int koff = (lane >> 4) << 4;
  const bf16_t* p = base + (size_t)m * stride + koff;
  BF16x16 u;
  u.h[0] = *(const v8bf*)(p);
  u.h[1] = *(const v8bf*)(p + 8);
  return u.v;
}

// B fragment 32x16 bf16, element (k,n) stored at base[n*stride + k]
// (k contiguous per lane -> contiguous 16B loads).
//   lane L: n = L&15, k = ((L>=16)?16:0) + j
DEV v16bf load_b32x16_contig(const bf16_t* base, int stride, int lane) {
  const int n = lane & 15;
  const int kb = (lane >> 4) << 4;
  const bf16_t* p = base + (size_t)n * stride + kb;
  BF16x16 u;
  u.h[0] = *(const v8bf*)(p);
  u.h[1] = *(const v8bf*)(p + 8);
  return u.v;
}

// B fragment 32x16 bf16, element (k,n) stored at base[k*stride + n]
// (row-of-K layout; per-element LDS gathers).
DEV v16bf load_b32x16_strided(const bf16_t* base, int stride, int lane) {
  const int n = lane & 15;
  const int kb = (lane >> 4) << 4;
  v16bf b;
#pragma unroll
  for (int j = 0; j < 16; ++j) b[j] = base[(size_t)(kb + j) * stride + n];
  return b;
}

// ---------------------------------------------------------------------------
// Kernel 1: per-head projection  Y[hb][s][e] = sum_d x[b][s][d] * W[h][d][e]
// grid (S/128, D/128, H*B), block 256 (8 waves: 4 along M, 2 along N)
// ---------------------------------------------------------------------------
__global__ __launch_bounds__(256)
void proj_kernel(const float* __restrict__ x, const float* __restrict__ W,
                 bf16_t* __restrict__ Y) {
  __shared__ bf16_t As[128][40];   // x tile  [m][k], bf16
  __shared__ bf16_t Bs[32][132];   // W tile  [k][n], bf16

  const int tid = threadIdx.x, lane = tid & 31, wave = tid >> 5;
  const int hb = blockIdx.z;                 // h*4 + b
  const int h = hb >> 2, b = hb & 3;
  const int mblk = blockIdx.x * 128;
  const int nblk = blockIdx.y * 128;
  const int wm = (wave & 3) * 32;
  const int wn = (wave >> 2) * 64;

  const float* xA = x + ((size_t)b * 4096 + mblk) * 256;
  const float* WB = W + (size_t)h * 65536 + nblk;

  v8f acc[2][4];
#pragma unroll
  for (int mi = 0; mi < 2; ++mi)
#pragma unroll
    for (int ni = 0; ni < 4; ++ni) acc[mi][ni] = vzero8();

  for (int k0 = 0; k0 < 256; k0 += 32) {
    // stage A: 128x32 f32 -> bf16 (coalesced float4 loads)
#pragma unroll
    for (int i = 0; i < 4; ++i) {
      const int linear = (i * 256 + tid) * 4;
      const int row = linear >> 5, col = linear & 31;
      v4f v = *(const v4f*)(xA + (size_t)row * 256 + k0 + col);
#pragma unroll
      for (int e = 0; e < 4; ++e) As[row][col + e] = (bf16_t)v[e];
    }
    // stage B: 32x128 f32 -> bf16
#pragma unroll
    for (int i = 0; i < 4; ++i) {
      const int linear = (i * 256 + tid) * 4;
      const int row = linear >> 7, col = linear & 127;
      v4f v = *(const v4f*)(WB + (size_t)(k0 + row) * 256 + col);
#pragma unroll
      for (int e = 0; e < 4; ++e) Bs[row][col + e] = (bf16_t)v[e];
    }
    __syncthreads();

    v16bf af0 = load_a16x32(&As[wm][0], 40, lane);
    v16bf af1 = load_a16x32(&As[wm + 16][0], 40, lane);
#pragma unroll
    for (int ni = 0; ni < 4; ++ni) {
      v16bf bfrag = load_b32x16_strided(&Bs[0][wn + ni * 16], 132, lane);
      acc[0][ni] = wmma_bf16(af0, bfrag, acc[0][ni]);
      acc[1][ni] = wmma_bf16(af1, bfrag, acc[1][ni]);
    }
    __syncthreads();
  }

  // epilogue: C layout lane L, vgpr r -> (m = r + ((L>=16)?8:0), n = L&15)
  bf16_t* Yp = Y + ((size_t)hb * 4096 + mblk) * 256 + nblk;
  const int half8 = (lane >> 4) << 3;
  const int nl = lane & 15;
#pragma unroll
  for (int mi = 0; mi < 2; ++mi)
#pragma unroll
    for (int ni = 0; ni < 4; ++ni) {
      const int col = wn + ni * 16 + nl;
#pragma unroll
      for (int r = 0; r < 8; ++r)
        Yp[(size_t)(wm + mi * 16 + half8 + r) * 256 + col] =
            (bf16_t)acc[mi][ni][r];
    }
}

// ---------------------------------------------------------------------------
// Kernel 2: flash attention per (h,b,g) chunk; chunk g attends chunk (g+1)%4.
// grid (C/128, G, H*B), block 256 = 8 waves; each wave owns 16 query rows.
// Writes concat layout cat[b][s][h*256 + d] (bf16).
// ---------------------------------------------------------------------------
__global__ __launch_bounds__(256)
void attn_kernel(const bf16_t* __restrict__ Q, const bf16_t* __restrict__ K,
                 bf16_t* __restrict__ cat) {
  __shared__ bf16_t kvs[32][264];      // 32-key KV tile, [key][d]
  __shared__ bf16_t ps[8][16][40];     // per-wave P repack tile (C->A layout)

  const int tid = threadIdx.x, lane = tid & 31, wave = tid >> 5;
  const int hb = blockIdx.z, h = hb >> 2, b = hb & 3;
  const int g = blockIdx.y;
  const int qblk = blockIdx.x * 128;
  const int gk = (g + 1) & 3;          // rolled KV chunk

  const bf16_t* Qbase =
      Q + ((size_t)hb * 4096 + g * 1024 + qblk + wave * 16) * 256;
  const bf16_t* Kbase = K + ((size_t)hb * 4096 + gk * 1024) * 256;

  // Q fragments resident: 16 rows x 256 d = 8 A-frags
  v16bf qa[8];
#pragma unroll
  for (int dsi = 0; dsi < 8; ++dsi)
    qa[dsi] = load_a16x32(Qbase + dsi * 32, 256, lane);

  v8f o[16];
#pragma unroll
  for (int f = 0; f < 16; ++f) o[f] = vzero8();
  float m_run[8], l_run[8];
#pragma unroll
  for (int r = 0; r < 8; ++r) { m_run[r] = -1e30f; l_run[r] = 0.0f; }

  const int half8 = (lane >> 4) << 3;
  const int nl = lane & 15;

  for (int kv0 = 0; kv0 < 1024; kv0 += 32) {
    __syncthreads();
    // stage KV 32x256 bf16 -> LDS (coalesced 16B loads); prefetch next tile
#pragma unroll
    for (int i = 0; i < 4; ++i) {
      const int linear = (i * 256 + tid) * 8;
      const int row = linear >> 8, col = linear & 255;
      *(v8bf*)&kvs[row][col] =
          *(const v8bf*)(Kbase + (size_t)(kv0 + row) * 256 + col);
      if (kv0 + 32 < 1024)
        __builtin_prefetch(Kbase + (size_t)(kv0 + 32 + row) * 256 + col, 0, 1);
    }
    __syncthreads();

    // S = Q * KV^T for 32 keys: two 16x16 f32 frags (keys 0..15, 16..31)
    v8f s0 = vzero8(), s1 = vzero8();
#pragma unroll
    for (int dsi = 0; dsi < 8; ++dsi) {
      v16bf b0 = load_b32x16_contig(&kvs[0][dsi * 32], 264, lane);
      v16bf b1 = load_b32x16_contig(&kvs[16][dsi * 32], 264, lane);
      s0 = wmma_bf16(qa[dsi], b0, s0);
      s1 = wmma_bf16(qa[dsi], b1, s1);
    }

    // online softmax (no logit scaling, faithful to reference)
    float ex0[8], ex1[8];
#pragma unroll
    for (int r = 0; r < 8; ++r) {
      float mx = fmaxf(s0[r], s1[r]);
      mx = fmaxf(mx, __shfl_xor(mx, 1, 32));
      mx = fmaxf(mx, __shfl_xor(mx, 2, 32));
      mx = fmaxf(mx, __shfl_xor(mx, 4, 32));
      mx = fmaxf(mx, __shfl_xor(mx, 8, 32));   // row max across 16-lane half
      const float mnew = fmaxf(m_run[r], mx);
      const float corr = __expf(m_run[r] - mnew);
      m_run[r] = mnew;
      const float p0 = __expf(s0[r] - mnew);
      const float p1 = __expf(s1[r] - mnew);
      float psum = p0 + p1;
      psum += __shfl_xor(psum, 1, 32);
      psum += __shfl_xor(psum, 2, 32);
      psum += __shfl_xor(psum, 4, 32);
      psum += __shfl_xor(psum, 8, 32);
      l_run[r] = l_run[r] * corr + psum;
#pragma unroll
      for (int f = 0; f < 16; ++f) o[f][r] *= corr;   // rescale O row r
      ex0[r] = p0; ex1[r] = p1;
    }

    // repack P: C-layout f32 -> A-layout bf16 via LDS
#pragma unroll
    for (int r = 0; r < 8; ++r) {
      ps[wave][half8 + r][nl]      = (bf16_t)ex0[r];
      ps[wave][half8 + r][16 + nl] = (bf16_t)ex1[r];
    }
    __syncthreads();

    // O += P(16x32) @ KV(32x256): 16 WMMAs over d-blocks
    v16bf pa = load_a16x32(&ps[wave][0][0], 40, lane);
#pragma unroll
    for (int nb = 0; nb < 16; ++nb) {
      v16bf bv = load_b32x16_strided(&kvs[0][nb * 16], 264, lane);
      o[nb] = wmma_bf16(pa, bv, o[nb]);
    }
  }

  // epilogue: normalize and scatter into concat-head layout
  const int s0row = g * 1024 + qblk + wave * 16;
  bf16_t* cp = cat + ((size_t)b * 4096 + s0row) * 2560 + h * 256;
#pragma unroll
  for (int r = 0; r < 8; ++r) {
    const float inv = 1.0f / l_run[r];
    const int m = half8 + r;
#pragma unroll
    for (int f = 0; f < 16; ++f)
      cp[(size_t)m * 2560 + f * 16 + nl] = (bf16_t)(o[f][r] * inv);
  }
}

// ---------------------------------------------------------------------------
// Kernel 3: out[m][n] = sum_k cat[m][k] * Wo[n][k]; M=16384, N=256, K=2560.
// grid (M/128, N/128), block 256 (8 waves: 4M x 2N). f32 output.
// ---------------------------------------------------------------------------
__global__ __launch_bounds__(256)
void out_kernel(const bf16_t* __restrict__ cat, const float* __restrict__ Wo,
                float* __restrict__ out) {
  __shared__ bf16_t As[128][40];   // cat tile [m][k]
  __shared__ bf16_t Ws[128][40];   // Wo  tile [n][k] (B^T in k-contig form)

  const int tid = threadIdx.x, lane = tid & 31, wave = tid >> 5;
  const int mblk = blockIdx.x * 128;
  const int nblk = blockIdx.y * 128;
  const int wm = (wave & 3) * 32;
  const int wn = (wave >> 2) * 64;

  v8f acc[2][4];
#pragma unroll
  for (int mi = 0; mi < 2; ++mi)
#pragma unroll
    for (int ni = 0; ni < 4; ++ni) acc[mi][ni] = vzero8();

  for (int k0 = 0; k0 < 2560; k0 += 32) {
    // stage A (already bf16): 128x32, 16B vector loads
#pragma unroll
    for (int i = 0; i < 2; ++i) {
      const int linear = (i * 256 + tid) * 8;
      const int row = linear >> 5, col = linear & 31;
      *(v8bf*)&As[row][col] =
          *(const v8bf*)(cat + (size_t)(mblk + row) * 2560 + k0 + col);
    }
    // stage B: Wo[n][k] f32 -> bf16, kept k-contiguous
#pragma unroll
    for (int i = 0; i < 4; ++i) {
      const int linear = (i * 256 + tid) * 4;
      const int row = linear >> 5, col = linear & 31;
      v4f v = *(const v4f*)(Wo + (size_t)(nblk + row) * 2560 + k0 + col);
#pragma unroll
      for (int e = 0; e < 4; ++e) Ws[row][col + e] = (bf16_t)v[e];
    }
    __syncthreads();

    v16bf af0 = load_a16x32(&As[wm][0], 40, lane);
    v16bf af1 = load_a16x32(&As[wm + 16][0], 40, lane);
#pragma unroll
    for (int ni = 0; ni < 4; ++ni) {
      v16bf bfrag = load_b32x16_contig(&Ws[wn + ni * 16][0], 40, lane);
      acc[0][ni] = wmma_bf16(af0, bfrag, acc[0][ni]);
      acc[1][ni] = wmma_bf16(af1, bfrag, acc[1][ni]);
    }
    __syncthreads();
  }

  float* op = out + (size_t)mblk * 256 + nblk;
  const int half8 = (lane >> 4) << 3;
  const int nl = lane & 15;
#pragma unroll
  for (int mi = 0; mi < 2; ++mi)
#pragma unroll
    for (int ni = 0; ni < 4; ++ni) {
      const int col = wn + ni * 16 + nl;
#pragma unroll
      for (int r = 0; r < 8; ++r)
        op[(size_t)(wm + mi * 16 + half8 + r) * 256 + col] = acc[mi][ni][r];
    }
}

// ---------------------------------------------------------------------------
extern "C" void kernel_launch(void* const* d_in, const int* in_sizes, int n_in,
                              void* d_out, int out_size, void* d_ws,
                              size_t ws_size, hipStream_t stream) {
  const float* x  = (const float*)d_in[0];   // [4,4096,256]
  const float* Wq = (const float*)d_in[1];   // [10,256,256]
  const float* Wk = (const float*)d_in[2];   // [10,256,256]
  const float* Wo = (const float*)d_in[3];   // [256,2560]
  float* out = (float*)d_out;                // [4,4096,256]

  const size_t QK = (size_t)10 * 4 * 4096 * 256;  // elems per Q/K buffer
  bf16_t* Qbf = (bf16_t*)d_ws;
  bf16_t* Kbf = Qbf + QK;
  bf16_t* cat = Kbf + QK;                    // [4,4096,2560] bf16

  dim3 blk(256);
  proj_kernel<<<dim3(32, 2, 40), blk, 0, stream>>>(x, Wq, Qbf);
  proj_kernel<<<dim3(32, 2, 40), blk, 0, stream>>>(x, Wk, Kbf);
  attn_kernel<<<dim3(8, 4, 40), blk, 0, stream>>>(Qbf, Kbf, cat);
  out_kernel<<<dim3(128, 2, 1), blk, 0, stream>>>(cat, Wo, out);
}
